// BaselineOrbitals_73040213836023
// MI455X (gfx1250) — compile-verified, matrix operand order checked by reading, not run on hardware
//
#include <hip/hip_runtime.h>
#include <hip/hip_bf16.h>
#include <math.h>

// Problem constants
#define B_    256
#define NE_   32
#define NU_   16
#define NI_   8
#define NDET_ 16
#define D_    256
#define DP_   32
#define HS_   128
#define HBF_  256

typedef __attribute__((ext_vector_type(16))) __bf16 v16bf;
typedef __attribute__((ext_vector_type(8)))  float  v8f;

__device__ inline __bf16 f2bf(float f) {
    union { float f; unsigned u; } x; x.f = f;
    unsigned r = x.u + 0x7fffu + ((x.u >> 16) & 1u);   // round-to-nearest-even
    union { unsigned short s; __bf16 b; } y; y.s = (unsigned short)(r >> 16);
    return y.b;
}

// Load a WMMA 16x16x32 bf16 operand fragment: two contiguous 16B chunks.
// Caller passes rowptr = base + row*ld + k0 + 8*hi  (16B aligned).
__device__ inline v16bf ld_frag(const __bf16* rowptr) {
    union { v16bf v; uint4 q[2]; } u;
    u.q[0] = *(const uint4*)(rowptr);
    u.q[1] = *(const uint4*)(rowptr + 16);
    return u.v;
}

// Same, from an LDS tile (row-major, row stride ld elements).
__device__ inline v16bf ld_op(const __bf16* tile, int ld) {
    int l  = threadIdx.x & 31;
    int r  = l & 15, hi = l >> 4;
    return ld_frag(tile + r * ld + 8 * hi);
}

__device__ inline v8f wmma_bf16(v16bf a, v16bf b, v8f c) {
    return __builtin_amdgcn_wmma_f32_16x16x32_bf16(false, a, false, b, (short)0, c, false, false);
}

// ---------------------------------------------------------------------------
// Conversion kernels: run once per launch, feed all WMMA stages.
// ---------------------------------------------------------------------------
__global__ void k_cvt_flat(const float* __restrict__ src, __bf16* __restrict__ dst, int n) {
    for (int i = blockIdx.x * blockDim.x + threadIdx.x; i < n; i += gridDim.x * blockDim.x)
        dst[i] = f2bf(src[i]);
}

// dst[n*K + k] = src[k*N + n]   (weights: row-major [K][N] -> packed B^T [N][K])
__global__ void k_cvt_t(const float* __restrict__ src, __bf16* __restrict__ dst, int K, int N) {
    int total = K * N;
    for (int i = blockIdx.x * blockDim.x + threadIdx.x; i < total; i += gridDim.x * blockDim.x) {
        int n = i / K, k = i - n * K;
        dst[i] = f2bf(src[k * N + n]);
    }
}

// ---------------------------------------------------------------------------
// K1: base[b,i,:] = emb_el[b,i,:] @ w0[:256,:] + b0 for both shift nets.
// One wave per 16-row tile; operands loaded straight from global bf16.
// ---------------------------------------------------------------------------
__global__ __launch_bounds__(32) void k_base(const __bf16* __restrict__ embb,
        const __bf16* __restrict__ w0Ta, const float* __restrict__ b0a,
        const __bf16* __restrict__ w0Tb, const float* __restrict__ b0b,
        float* __restrict__ basea, float* __restrict__ baseb) {
    const __bf16* w0T = blockIdx.y ? w0Tb : w0Ta;
    const float*  b0  = blockIdx.y ? b0b  : b0a;
    float* base       = blockIdx.y ? baseb : basea;

    int t0 = blockIdx.x * 16;
    int l = threadIdx.x, nl = l & 15, hi = l >> 4;
    const __bf16* arow = embb + (t0 + nl) * 256 + 8 * hi;

    v8f acc[8];
    for (int nt = 0; nt < 8; ++nt)
        for (int r = 0; r < 8; ++r) acc[nt][r] = 0.f;

    for (int k0 = 0; k0 < 256; k0 += 32) {
        v16bf a = ld_frag(arow + k0);
        for (int nt = 0; nt < 8; ++nt) {
            v16bf b = ld_frag(w0T + (nt * 16 + nl) * 256 + k0 + 8 * hi);
            acc[nt] = wmma_bf16(a, b, acc[nt]);
        }
    }
    for (int nt = 0; nt < 8; ++nt) {
        float bias = b0[nt * 16 + nl];
        for (int r = 0; r < 8; ++r)
            base[(t0 + r + 8 * hi) * HS_ + nt * 16 + nl] = acc[nt][r] + bias;
    }
}

// ---------------------------------------------------------------------------
// K2: pair part.  h = base[row/NP] + pair_emb @ w0[256:288,:];
// s = tanh(h) @ w1;  atomic accumulate s/(1+d^3)*diff into out[b,i,:].
// One wave per 16 pair rows; single K=32 WMMA step, operands from global bf16.
// ---------------------------------------------------------------------------
__global__ __launch_bounds__(32) void k_pair(int NP,
        const __bf16* __restrict__ pairb, const float* __restrict__ diff,
        const float* __restrict__ dist,
        const __bf16* __restrict__ wT, const float* __restrict__ w1,
        const float* __restrict__ base, float* __restrict__ out) {
    int t0 = blockIdx.x * 16;
    int l = threadIdx.x, nl = l & 15, hi = l >> 4;

    v8f acc[8];
    for (int nt = 0; nt < 8; ++nt)
        for (int r = 0; r < 8; ++r)
            acc[nt][r] = base[((t0 + r + 8 * hi) / NP) * HS_ + nt * 16 + nl];

    v16bf a = ld_frag(pairb + (t0 + nl) * 32 + 8 * hi);
    for (int nt = 0; nt < 8; ++nt) {
        v16bf b = ld_frag(wT + (nt * 16 + nl) * 32 + 8 * hi);
        acc[nt] = wmma_bf16(a, b, acc[nt]);
    }

    float t[8];
    for (int r = 0; r < 8; ++r) t[r] = 0.f;
    for (int nt = 0; nt < 8; ++nt) {
        float w = w1[nt * 16 + nl];
        for (int r = 0; r < 8; ++r) t[r] += tanhf(acc[nt][r]) * w;
    }
    for (int m = 1; m < 16; m <<= 1)
        for (int r = 0; r < 8; ++r) t[r] += __shfl_xor(t[r], m, 32);

    if (nl == 0) {   // lanes 0 and 16 each own rows m = r + 8*hi
        for (int r = 0; r < 8; ++r) {
            int   row = t0 + r + 8 * hi;
            float d   = dist[row];
            float w   = t[r] / (1.f + d * d * d);
            int   ig  = row / NP;   // = b*NE + i
            for (int dd = 0; dd < 3; ++dd)
                atomicAdd(&out[ig * 3 + dd], w * diff[row * 3 + dd]);
        }
    }
}

// ---------------------------------------------------------------------------
// K4: decay + shift + dist_ei^2 -> AO -> MO -> gather(ind_orb) -> CI scale,
// scatter into mo_ws[b, det, e, 32].  One thread per (b, e).
// ---------------------------------------------------------------------------
__global__ __launch_bounds__(256) void k_mo(
        const float* __restrict__ se, const float* __restrict__ si,
        const float* __restrict__ diff_ei, const float* __restrict__ dist_ei,
        const float* __restrict__ dscale, const float* __restrict__ Z,
        const float* __restrict__ alpha,
        const float* __restrict__ coef_up, const float* __restrict__ coef_dn,
        const float* __restrict__ ci,
        const int* __restrict__ ind_up, const int* __restrict__ ind_dn,
        float* __restrict__ mo_ws) {
    int tid = blockIdx.x * blockDim.x + threadIdx.x;   // = b*NE + e
    int b = tid >> 5, e = tid & 31;

    float ds = dscale[0];
    float decay = 1.f;
    for (int a = 0; a < NI_; ++a) {
        float q = dist_ei[tid * NI_ + a] * Z[a] / ds;
        decay *= tanhf(q * q);
    }
    float sh[3];
    for (int d = 0; d < 3; ++d)
        sh[d] = (se[tid * 3 + d] + si[tid * 3 + d]) * decay;

    const float* coef = (e < NU_) ? coef_up : coef_dn;
    float acc[48];
    for (int n = 0; n < 48; ++n) acc[n] = 0.f;
    for (int a = 0; a < NI_; ++a) {
        float r2 = 0.f;
        for (int d = 0; d < 3; ++d) {
            float de = diff_ei[(tid * NI_ + a) * 3 + d] + sh[d];
            r2 += de * de;
        }
        for (int k = 0; k < 16; ++k) {
            float ao = expf(-alpha[a * 16 + k] * r2);
            const float* crow = coef + (a * 16 + k) * 48;
            for (int n = 0; n < 48; ++n) acc[n] += ao * crow[n];
        }
    }

    for (int det = 0; det < NDET_; ++det) {
        float cw  = ci[det];
        float mag = powf(fabsf(cw), 1.f / 16.f);
        float sgn = (cw > 0.f) ? 1.f : ((cw < 0.f) ? -1.f : 0.f);
        float* dst = mo_ws + ((b * NDET_ + det) * NE_ + e) * 32;
        if (e < NU_) {
            for (int c = 0; c < 16; ++c)
                dst[c] = acc[ind_up[det * 16 + c]] * mag * (c == 0 ? sgn : 1.f);
            for (int c = 16; c < 32; ++c) dst[c] = 0.f;
        } else {
            for (int c = 0; c < 16; ++c) dst[c] = 0.f;
            for (int c = 16; c < 32; ++c)
                dst[c] = acc[ind_dn[det * 16 + (c - 16)]];
        }
    }
}

// ---------------------------------------------------------------------------
// K5: backflow MLP (256 -> tanh 256 -> 512) via two chained WMMA GEMMs,
// then multiply with gathered MO and write final output.
// 4 waves/block, one 16-row tile; only 8KB LDS for the hidden layer.
// ---------------------------------------------------------------------------
__global__ __launch_bounds__(128) void k_bf(
        const __bf16* __restrict__ embb,
        const __bf16* __restrict__ upw0T, const float* __restrict__ up_b0,
        const __bf16* __restrict__ upw1T, const float* __restrict__ up_b1,
        const __bf16* __restrict__ dnw0T, const float* __restrict__ dn_b0,
        const __bf16* __restrict__ dnw1T, const float* __restrict__ dn_b1,
        const float* __restrict__ mo_ws, float* __restrict__ out) {
    __shared__ __attribute__((aligned(16))) __bf16 hT[16 * 256];

    int t0 = blockIdx.x * 16;     // global (b,e) row
    int b  = t0 >> 5;
    int e0 = t0 & 31;             // 0 or 16
    bool is_up = (e0 == 0);
    const __bf16* w0T = is_up ? upw0T : dnw0T;
    const float*  b0  = is_up ? up_b0 : dn_b0;
    const __bf16* w1T = is_up ? upw1T : dnw1T;
    const float*  b1  = is_up ? up_b1 : dn_b1;

    int t = threadIdx.x;
    int wid = t >> 5, l = t & 31, nl = l & 15, hi = l >> 4;
    const __bf16* arow = embb + (t0 + nl) * 256 + 8 * hi;

    // GEMM1: N=256, each wave owns 4 n-tiles
    v8f acc1[4];
    for (int nt = 0; nt < 4; ++nt) {
        float bias = b0[(wid * 4 + nt) * 16 + nl];
        for (int r = 0; r < 8; ++r) acc1[nt][r] = bias;
    }
    for (int k0 = 0; k0 < 256; k0 += 32) {
        v16bf a = ld_frag(arow + k0);
        for (int nt = 0; nt < 4; ++nt) {
            int n = (wid * 4 + nt) * 16 + nl;
            acc1[nt] = wmma_bf16(a, ld_frag(w0T + n * 256 + k0 + 8 * hi), acc1[nt]);
        }
    }
    for (int nt = 0; nt < 4; ++nt) {
        int n = (wid * 4 + nt) * 16 + nl;
        for (int r = 0; r < 8; ++r)
            hT[(r + 8 * hi) * 256 + n] = f2bf(tanhf(acc1[nt][r]));
    }
    __syncthreads();

    // GEMM2: N=512, each wave owns 8 n-tiles
    v8f acc2[8];
    for (int nt = 0; nt < 8; ++nt) {
        float bias = b1[(wid * 8 + nt) * 16 + nl];
        for (int r = 0; r < 8; ++r) acc2[nt][r] = bias;
    }
    for (int k0 = 0; k0 < 256; k0 += 32) {
        v16bf a = ld_op(hT + k0, 256);
        for (int nt = 0; nt < 8; ++nt) {
            int n = (wid * 8 + nt) * 16 + nl;
            acc2[nt] = wmma_bf16(a, ld_frag(w1T + n * 256 + k0 + 8 * hi), acc2[nt]);
        }
    }

    const int HALF = B_ * NDET_ * 16 * 32;
    for (int nt = 0; nt < 8; ++nt) {
        int ycol = (wid * 8 + nt) * 16 + nl;
        int det = ycol >> 5, c = ycol & 31;
        for (int r = 0; r < 8; ++r) {
            int m = r + 8 * hi;
            int e = e0 + m;
            float v = acc2[nt][r] * mo_ws[((b * NDET_ + det) * NE_ + e) * 32 + c];
            int eo = is_up ? e : (e - 16);
            float* dst = out + (is_up ? 0 : HALF);
            dst[((b * NDET_ + det) * 16 + eo) * 32 + c] = v;
        }
    }
}

// ---------------------------------------------------------------------------
extern "C" void kernel_launch(void* const* d_in, const int* in_sizes, int n_in,
                              void* d_out, int out_size, void* d_ws, size_t ws_size,
                              hipStream_t stream) {
    const float* diff_ee = (const float*)d_in[0];
    const float* dist_ee = (const float*)d_in[1];
    const float* diff_ei = (const float*)d_in[2];
    const float* dist_ei = (const float*)d_in[3];
    const float* emb_el  = (const float*)d_in[4];
    const float* emb_ee  = (const float*)d_in[5];
    const float* emb_ei  = (const float*)d_in[6];
    const float* sel_w0  = (const float*)d_in[7];
    const float* sel_b0  = (const float*)d_in[8];
    const float* sel_w1  = (const float*)d_in[9];
    const float* sio_w0  = (const float*)d_in[10];
    const float* sio_b0  = (const float*)d_in[11];
    const float* sio_w1  = (const float*)d_in[12];
    const float* up_w0   = (const float*)d_in[13];
    const float* up_b0   = (const float*)d_in[14];
    const float* up_w1   = (const float*)d_in[15];
    const float* up_b1   = (const float*)d_in[16];
    const float* dn_w0   = (const float*)d_in[17];
    const float* dn_b0   = (const float*)d_in[18];
    const float* dn_w1   = (const float*)d_in[19];
    const float* dn_b1   = (const float*)d_in[20];
    const float* dscale  = (const float*)d_in[21];
    const float* Z       = (const float*)d_in[22];
    const float* alpha   = (const float*)d_in[23];
    const float* coef_up = (const float*)d_in[24];
    const float* coef_dn = (const float*)d_in[25];
    const float* ciw     = (const float*)d_in[26];
    const int*   ind_up  = (const int*)d_in[27];
    const int*   ind_dn  = (const int*)d_in[28];

    char* ws = (char*)d_ws;
    size_t o = 0;
    float* se       = (float*)(ws + o); o += (size_t)B_*NE_*3 * 4;          // 98304
    float* si       = (float*)(ws + o); o += (size_t)B_*NE_*3 * 4;
    float* base_el  = (float*)(ws + o); o += (size_t)B_*NE_*HS_ * 4;        // 4MB
    float* base_io  = (float*)(ws + o); o += (size_t)B_*NE_*HS_ * 4;
    __bf16* emb_bf  = (__bf16*)(ws + o); o += (size_t)B_*NE_*D_ * 2;        // 4MB
    __bf16* embei_bf= (__bf16*)(ws + o); o += (size_t)B_*NE_*NI_*DP_ * 2;   // 4MB
    __bf16* w0Tel   = (__bf16*)(ws + o); o += (size_t)HS_*256 * 2;
    __bf16* w0Tio   = (__bf16*)(ws + o); o += (size_t)HS_*256 * 2;
    __bf16* w0tTel  = (__bf16*)(ws + o); o += (size_t)HS_*32 * 2;
    __bf16* w0tTio  = (__bf16*)(ws + o); o += (size_t)HS_*32 * 2;
    __bf16* upw0T   = (__bf16*)(ws + o); o += (size_t)256*256 * 2;
    __bf16* dnw0T   = (__bf16*)(ws + o); o += (size_t)256*256 * 2;
    __bf16* upw1T   = (__bf16*)(ws + o); o += (size_t)512*256 * 2;
    __bf16* dnw1T   = (__bf16*)(ws + o); o += (size_t)512*256 * 2;
    // Aliased region: embee_bf (16.78MB, dead after k_pair) == mo_ws (16.78MB)
    __bf16* embee_bf = (__bf16*)(ws + o);
    float*  mo_ws    = (float*)(ws + o);

    hipMemsetAsync(se, 0, (size_t)2 * B_*NE_*3 * sizeof(float), stream);

    // One-time conversions / packing (bf16; weights transposed to [N][K]).
    k_cvt_flat<<<1024, 256, 0, stream>>>(emb_el, emb_bf, B_*NE_*D_);
    k_cvt_flat<<<2048, 256, 0, stream>>>(emb_ee, embee_bf, B_*NE_*NE_*DP_);
    k_cvt_flat<<<1024, 256, 0, stream>>>(emb_ei, embei_bf, B_*NE_*NI_*DP_);
    k_cvt_t<<<128, 256, 0, stream>>>(sel_w0,            w0Tel,  256, HS_);
    k_cvt_t<<<128, 256, 0, stream>>>(sio_w0,            w0Tio,  256, HS_);
    k_cvt_t<<<16,  256, 0, stream>>>(sel_w0 + 256*HS_,  w0tTel, 32,  HS_);
    k_cvt_t<<<16,  256, 0, stream>>>(sio_w0 + 256*HS_,  w0tTio, 32,  HS_);
    k_cvt_t<<<256, 256, 0, stream>>>(up_w0,             upw0T,  256, 256);
    k_cvt_t<<<256, 256, 0, stream>>>(dn_w0,             dnw0T,  256, 256);
    k_cvt_t<<<512, 256, 0, stream>>>(up_w1,             upw1T,  256, 512);
    k_cvt_t<<<512, 256, 0, stream>>>(dn_w1,             dnw1T,  256, 512);

    // base GEMMs for both shift nets (N=128, K=256)
    k_base<<<dim3((B_*NE_)/16, 2), 32, 0, stream>>>(emb_bf, w0Tel, sel_b0,
                                                    w0Tio, sio_b0, base_el, base_io);
    // pair parts: el-el (NP=32) and el-ion (NP=8)
    k_pair<<<(B_*NE_*NE_)/16, 32, 0, stream>>>(NE_, embee_bf, diff_ee, dist_ee,
                                               w0tTel, sel_w1, base_el, se);
    k_pair<<<(B_*NE_*NI_)/16, 32, 0, stream>>>(NI_, embei_bf, diff_ei, dist_ei,
                                               w0tTio, sio_w1, base_io, si);
    // decay + shift + AO/MO + gather + CI   (writes mo_ws, aliases embee_bf)
    k_mo<<<(B_*NE_)/256, 256, 0, stream>>>(se, si, diff_ei, dist_ei, dscale, Z,
                                           alpha, coef_up, coef_dn, ciw,
                                           ind_up, ind_dn, mo_ws);
    // backflow MLPs + final multiply
    k_bf<<<(B_*NE_)/16, 128, 0, stream>>>(emb_bf, upw0T, up_b0, upw1T, up_b1,
                                          dnw0T, dn_b0, dnw1T, dn_b1,
                                          mo_ws, (float*)d_out);
}
